// TurboQuantLinear_40870908788960
// MI455X (gfx1250) — compile-verified, matrix operand order checked by reading, not run on hardware
//
#include <hip/hip_runtime.h>
#include <hip/hip_bf16.h>
#include <stdint.h>

// Problem constants (match reference)
#define M_TOK 4096
#define N_OUT 11008
#define K_IN  4096
#define GRP   128
#define NGRP  32

typedef __attribute__((ext_vector_type(16))) __bf16 v16bf;
typedef __attribute__((ext_vector_type(8)))  __bf16 v8bf;
typedef __attribute__((ext_vector_type(8)))  float  v8f;

// ---- round-to-nearest-even fp32 -> bf16 bits --------------------------------
__device__ __forceinline__ unsigned short f32_to_bf16_rne(float f) {
    union { float f; unsigned u; } c; c.f = f;
    unsigned u = c.u;
    u += 0x7FFFu + ((u >> 16) & 1u);
    return (unsigned short)(u >> 16);
}

// ---- kernel 1: x fp32 -> bf16 ----------------------------------------------
__global__ __launch_bounds__(256) void cvt_x_kernel(const float* __restrict__ x,
                                                    unsigned short* __restrict__ Xb) {
    int i = (blockIdx.x * 256 + threadIdx.x) * 4;   // 4 floats / thread
    float4 f = *(const float4*)&x[i];
    unsigned lo = (unsigned)f32_to_bf16_rne(f.x) | ((unsigned)f32_to_bf16_rne(f.y) << 16);
    unsigned hi = (unsigned)f32_to_bf16_rne(f.z) | ((unsigned)f32_to_bf16_rne(f.w) << 16);
    uint2 v; v.x = lo; v.y = hi;
    *(uint2*)&Xb[i] = v;
}

// ---- kernel 2: dequantize W to bf16 ----------------------------------------
// One 128-thread block per output row o. Rotation (128x128 f32 = 64KB) staged
// in LDS once per block (CDNA5 WGP has 320KB LDS). For each of the 32 groups:
// decode 128 4-bit indices -> codebook values v[], then u[h] = dot(v, R[h,:]),
// w = u * norms[o,g], stored bf16.
__global__ __launch_bounds__(128) void dequant_kernel(const int* __restrict__ packed,
                                                      const float* __restrict__ codebook,
                                                      const float* __restrict__ norms,
                                                      const float* __restrict__ rot,
                                                      unsigned short* __restrict__ Wb) {
    __shared__ float rots[GRP * GRP];   // 64 KB
    __shared__ float vsh[GRP];
    __shared__ float cb[16];
    const int t = threadIdx.x;
    const int o = blockIdx.x;

    if (t < 16) cb[t] = codebook[t];
    #pragma unroll 4
    for (int i = t * 4; i < GRP * GRP; i += 128 * 4)
        *(float4*)&rots[i] = *(const float4*)&rot[i];
    __syncthreads();

    for (int g = 0; g < NGRP; ++g) {
        if (t < 64) {
            int p = packed[(size_t)o * (K_IN / 2) + g * 64 + t];
            vsh[2 * t]     = cb[p & 15];
            vsh[2 * t + 1] = cb[(p >> 4) & 15];
        }
        __syncthreads();
        const float4* vr = (const float4*)vsh;
        const float4* rr = (const float4*)&rots[t * GRP];
        float s = 0.f;
        #pragma unroll
        for (int i = 0; i < GRP / 4; ++i) {
            float4 a = rr[i], b = vr[i];
            s = fmaf(a.x, b.x, fmaf(a.y, b.y, fmaf(a.z, b.z, fmaf(a.w, b.w, s))));
        }
        float w = s * norms[(size_t)o * NGRP + g];
        Wb[(size_t)o * K_IN + g * GRP + t] = f32_to_bf16_rne(w);
        __syncthreads();   // protect vsh before next group's decode
    }
}

// ---- fragment load helper (ISA 16-bit A/B operand layout) -------------------
// lanes 0-15 : row = lane,    chunks K[0..7]  (VGPR0-3) and K[16..23] (VGPR4-7)
// lanes 16-31: row = lane-16, chunks K[8..15] (VGPR0-3) and K[24..31] (VGPR4-7)
__device__ __forceinline__ v16bf load_frag(const unsigned short* __restrict__ base, int k0) {
    v8bf lo = *(const v8bf*)(base + k0);        // ds_load_b128
    v8bf hi = *(const v8bf*)(base + k0 + 16);   // ds_load_b128
    v16bf r;
    #pragma unroll
    for (int i = 0; i < 8; ++i) { r[i] = lo[i]; r[8 + i] = hi[i]; }
    return r;
}

// ---- kernel 3: WMMA bf16 GEMM: out[M,N] = Xb[M,K] * Wb[N,K]^T + bias --------
// 256 threads = 8 waves; block tile 128(M) x 128(N), BK = 32.
// Wave (wm,wn) in a 2x4 grid owns a 64x32 sub-tile = 4x2 fragments of 16x16.
// Software-pipelined: next K-chunk held in registers while current chunk is
// consumed from LDS; ONE barrier per K-step (double-buffered LDS makes it safe:
// a wave stores into buf[k&1] only after barrier(k-1), which all waves pass
// only after finishing compute(k-2), the last reader of that buffer).
__global__ __launch_bounds__(256) void gemm_wmma_kernel(const unsigned short* __restrict__ Xb,
                                                        const unsigned short* __restrict__ Wb,
                                                        const float* __restrict__ bias,
                                                        float* __restrict__ out) {
    __shared__ unsigned short Xs[2][128 * 32];   // 2 x 8 KB
    __shared__ unsigned short Ws[2][128 * 32];   // 2 x 8 KB

    const int t    = threadIdx.x;
    const int lane = t & 31;
    const int wave = t >> 5;
    const int wm   = wave & 1;        // 0..1 -> M offset wm*64
    const int wn   = wave >> 1;       // 0..3 -> N offset wn*32
    const int half = lane >> 4;       // lane half select
    const int r16  = lane & 15;
    const int k0   = half * 8;

    const int bm = blockIdx.y * 128;
    const int bn = blockIdx.x * 128;

    v8f acc[4][2];
    #pragma unroll
    for (int i = 0; i < 4; ++i)
        #pragma unroll
        for (int j = 0; j < 2; ++j)
            acc[i][j] = (v8f)(0.f);

    // Each thread stages two 16-byte chunks per tile (128x32 bf16 = 8 KB;
    // 256 threads x 2 x 16 B).
    const int e0 = t * 8;           // rows 0..63
    const int e1 = 2048 + t * 8;    // rows 64..127
    const int row0 = e0 >> 5, col0 = e0 & 31;
    const int row1 = e1 >> 5, col1 = e1 & 31;

    const unsigned short* gx0 = &Xb[(size_t)(bm + row0) * K_IN + col0];
    const unsigned short* gx1 = &Xb[(size_t)(bm + row1) * K_IN + col1];
    const unsigned short* gw0 = &Wb[(size_t)(bn + row0) * K_IN + col0];
    const unsigned short* gw1 = &Wb[(size_t)(bn + row1) * K_IN + col1];

    // Prologue: fetch K-chunk 0 into registers.
    uint4 rx0 = *(const uint4*)gx0;
    uint4 rx1 = *(const uint4*)gx1;
    uint4 rw0 = *(const uint4*)gw0;
    uint4 rw1 = *(const uint4*)gw1;

    const int NK = K_IN / 32;       // 128 K-steps
    for (int k = 0; k < NK; ++k) {
        const int p = k & 1;
        // Commit in-flight registers to the current LDS buffer.
        *(uint4*)&Xs[p][e0] = rx0;
        *(uint4*)&Xs[p][e1] = rx1;
        *(uint4*)&Ws[p][e0] = rw0;
        *(uint4*)&Ws[p][e1] = rw1;
        __syncthreads();

        // Issue next chunk's global loads; they complete under the WMMAs below
        // (the s_wait_loadcnt lands before next iteration's ds_store).
        if (k + 1 < NK) {
            const int off = (k + 1) * 32;
            rx0 = *(const uint4*)(gx0 + off);
            rx1 = *(const uint4*)(gx1 + off);
            rw0 = *(const uint4*)(gw0 + off);
            rw1 = *(const uint4*)(gw1 + off);
            if (k + 2 < NK) {   // global_prefetch_b8: warm L2 one step further out
                __builtin_prefetch(gx0 + off + 32, 0, 0);
                __builtin_prefetch(gw0 + off + 32, 0, 0);
            }
        }

        // B fragments (N-major rows of Ws act as WMMA B operand)
        v16bf bf0 = load_frag(&Ws[p][(wn * 32 + 0  + r16) * 32], k0);
        v16bf bf1 = load_frag(&Ws[p][(wn * 32 + 16 + r16) * 32], k0);

        #pragma unroll
        for (int i = 0; i < 4; ++i) {
            v16bf a = load_frag(&Xs[p][(wm * 64 + i * 16 + r16) * 32], k0);
            acc[i][0] = __builtin_amdgcn_wmma_f32_16x16x32_bf16(
                false, a, false, bf0, (short)0, acc[i][0], false, false);
            acc[i][1] = __builtin_amdgcn_wmma_f32_16x16x32_bf16(
                false, a, false, bf1, (short)0, acc[i][1], false, false);
        }
    }

    // Epilogue: C/D layout — VGPR e holds row (e | half*8), column = lane&15
    #pragma unroll
    for (int i = 0; i < 4; ++i) {
        #pragma unroll
        for (int j = 0; j < 2; ++j) {
            const int gn = bn + wn * 32 + j * 16 + r16;
            const float bv = bias[gn];
            #pragma unroll
            for (int e = 0; e < 8; ++e) {
                const int gm = bm + wm * 64 + i * 16 + half * 8 + e;
                out[(size_t)gm * N_OUT + gn] = acc[i][j][e] + bv;
            }
        }
    }
}

// ---- launch -----------------------------------------------------------------
extern "C" void kernel_launch(void* const* d_in, const int* in_sizes, int n_in,
                              void* d_out, int out_size, void* d_ws, size_t ws_size,
                              hipStream_t stream) {
    const float* x        = (const float*)d_in[0];
    const int*   packed   = (const int*)d_in[1];
    const float* codebook = (const float*)d_in[2];
    const float* norms    = (const float*)d_in[3];
    const float* rot      = (const float*)d_in[4];
    const float* bias     = (const float*)d_in[5];
    float* out = (float*)d_out;

    unsigned short* Xb = (unsigned short*)d_ws;                                     // 33.5 MB
    unsigned short* Wb = (unsigned short*)((char*)d_ws + (size_t)M_TOK * K_IN * 2); // 90.2 MB

    // 1) x -> bf16
    cvt_x_kernel<<<(M_TOK * K_IN) / (256 * 4), 256, 0, stream>>>(x, Xb);
    // 2) dequantize W -> bf16
    dequant_kernel<<<N_OUT, 128, 0, stream>>>(packed, codebook, norms, rot, Wb);
    // 3) WMMA GEMM + bias
    dim3 grid(N_OUT / 128, M_TOK / 128);
    gemm_wmma_kernel<<<grid, 256, 0, stream>>>(Xb, Wb, bias, out);
}
